// DecoderBlock_72533407695477
// MI455X (gfx1250) — matrix operand order repaired and checked by reading, not compiled, not run on hardware
//
#include <hip/hip_runtime.h>

// Problem constants (match reference)
#define B_    4
#define S_    4096
#define E_    2048
#define H_    16
#define DH_   128
#define NSEG_ 8
#define SEG_  512
#define FE_   8192   // 4*E

typedef __attribute__((ext_vector_type(16))) __bf16          v16bf;
typedef __attribute__((ext_vector_type(16))) unsigned short  v16u;
typedef __attribute__((ext_vector_type(8)))  unsigned short  v8u;
typedef __attribute__((ext_vector_type(8)))  float           v8f;
typedef __attribute__((ext_vector_type(2)))  float           v2f;
typedef __attribute__((ext_vector_type(4)))  unsigned int    uint4v;
typedef __attribute__((ext_vector_type(8)))  unsigned int    uint8v;

__device__ __forceinline__ unsigned short f2bf(float f) {
  unsigned int u = __float_as_uint(f);
  unsigned int r = (u >> 16) & 1u;
  u += 0x7fffu + r;               // round-to-nearest-even
  return (unsigned short)(u >> 16);
}

// gfx1250 async copy: global memory -> LDS, tracked on ASYNCcnt.
__device__ __forceinline__ void async_g2l_b128(void* lds, const void* gaddr) {
  unsigned l = (unsigned)(unsigned long long)lds;    // low 32 bits = LDS offset
  asm volatile("global_load_async_to_lds_b128 %0, %1, off"
               :: "v"(l), "v"(gaddr) : "memory");
}
__device__ __forceinline__ void wait_asynccnt0() {
  asm volatile("s_wait_asynccnt 0x0" ::: "memory");
}

// ---------------------------------------------------------------- cast kernel
__global__ __launch_bounds__(256)
void k_cast_bf16(const float* __restrict__ src, unsigned short* __restrict__ dst,
                 long long n) {
  long long i = (long long)blockIdx.x * blockDim.x + threadIdx.x;
  long long stride = (long long)gridDim.x * blockDim.x;
  for (; i < n; i += stride) dst[i] = f2bf(src[i]);
}

// ---------------------------------------------------------------- bf16 GEMM
// C[M,N] f32 = A[M,K](bf16) * Bw[N,K](bf16)^T + bias[N]
// A tile staged via global_load_async_to_lds_b128 (ASYNCcnt),
// B (weight) tile staged via Tensor Data Mover (TENSORcnt), double-buffered.
#define TM 128
#define TN 128
#define TK 32

__global__ __launch_bounds__(256)
void k_gemm_bias(const unsigned short* __restrict__ A,
                 const unsigned short* __restrict__ Bw,
                 const float* __restrict__ bias,
                 float* __restrict__ C,
                 int Mm, int Nn, int Kk) {
  __shared__ __attribute__((aligned(16))) unsigned short As[2][TM][TK];
  __shared__ __attribute__((aligned(16))) unsigned short Bs[2][TN][TK];
  const int tid  = threadIdx.x;
  const int lane = tid & 31, wid = tid >> 5;
  const int half = lane >> 4, l16 = lane & 15;
  const int wm = wid & 3;            // 4 row bands of 32
  const int wn = wid >> 2;           // 2 col bands of 64
  const int bm = blockIdx.y * TM;
  const int bn = blockIdx.x * TN;
  const int lr = tid >> 1;           // 0..127
  const int lc = (tid & 1) << 4;     // 0 / 16

  // per-thread async staging of the A (activation) tile
  auto stageA = [&](int buf, int k0) {
    const unsigned short* g = &A[(size_t)(bm + lr) * Kk + k0 + lc];
    async_g2l_b128(&As[buf][lr][lc],     g);
    async_g2l_b128(&As[buf][lr][lc + 8], g + 8);
  };
  // one TDM descriptor per workgroup for the B (weight) tile: 128 x 32 bf16,
  // row stride Kk elements. Issued by wave 0 only; waited on TENSORcnt.
  auto stageB = [&](int buf, int k0) {
    if (wid != 0) return;
    unsigned lds_addr = (unsigned)(unsigned long long)&Bs[buf][0][0];
    unsigned long long ga =
        (unsigned long long)&Bw[(size_t)bn * Kk + k0];
    uint4v g0;
    g0[0] = 1u;                                            // count=1, user mode
    g0[1] = lds_addr;                                      // D#.lds_addr
    g0[2] = (unsigned)ga;                                  // global_addr[31:0]
    g0[3] = ((unsigned)(ga >> 32) & 0x01ffffffu) | (2u << 30);  // addr[56:32] | type=2
    uint8v g1;
    g1[0] = 1u << 16;                                      // data_size=1 (2 bytes)
    g1[1] = ((unsigned)Kk & 0xffffu) << 16;                // tensor_dim0 lo16
    g1[2] = ((unsigned)Kk >> 16) | (128u << 16);           // dim0 hi16 | tensor_dim1 lo16
    g1[3] = ((unsigned)TK) << 16;                          // tile_dim0 = 32
    g1[4] = 128u;                                          // tile_dim1 = 128
    g1[5] = (unsigned)Kk;                                  // tensor_dim0_stride lo32
    g1[6] = 0u;                                            // stride hi | dim1_stride lo
    g1[7] = 0u;
    asm volatile("tensor_load_to_lds %0, %1" :: "s"(g0), "s"(g1) : "memory");
  };

  const v8f vzero = {0.f,0.f,0.f,0.f,0.f,0.f,0.f,0.f};
  v8f acc[2][4];
  for (int i = 0; i < 2; ++i) for (int j = 0; j < 4; ++j) acc[i][j] = vzero;

  // prologue: fill buffer 0
  stageA(0, 0);
  stageB(0, 0);
  wait_asynccnt0();
  if (wid == 0) __builtin_amdgcn_s_wait_tensorcnt(0);
  __syncthreads();

  for (int k0 = 0; k0 < Kk; k0 += TK) {
    const int cur = (k0 >> 5) & 1;
    if (k0 + TK < Kk) {               // pipeline next tile behind the WMMAs
      stageA(cur ^ 1, k0 + TK);
      stageB(cur ^ 1, k0 + TK);
    }

    v16bf af[2], bfv[4];
    for (int mf = 0; mf < 2; ++mf) {
      const int m = wm * 32 + mf * 16 + l16;
      v8u lo = *(const v8u*)&As[cur][m][half * 8];
      v8u hi = *(const v8u*)&As[cur][m][16 + half * 8];
      v16u t;
      for (int i = 0; i < 8; ++i) { t[i] = lo[i]; t[i + 8] = hi[i]; }
      af[mf] = __builtin_bit_cast(v16bf, t);
    }
    for (int nf = 0; nf < 4; ++nf) {
      const int n = wn * 64 + nf * 16 + l16;
      v8u lo = *(const v8u*)&Bs[cur][n][half * 8];
      v8u hi = *(const v8u*)&Bs[cur][n][16 + half * 8];
      v16u t;
      for (int i = 0; i < 8; ++i) { t[i] = lo[i]; t[i + 8] = hi[i]; }
      bfv[nf] = __builtin_bit_cast(v16bf, t);
    }
    for (int mf = 0; mf < 2; ++mf)
      for (int nf = 0; nf < 4; ++nf)
        acc[mf][nf] = __builtin_amdgcn_wmma_f32_16x16x32_bf16(
            false, af[mf], false, bfv[nf], (short)0, acc[mf][nf], false, false);

    wait_asynccnt0();                               // next tile copies landed
    if (wid == 0) __builtin_amdgcn_s_wait_tensorcnt(0);
    __syncthreads();
  }

  for (int mf = 0; mf < 2; ++mf)
    for (int nf = 0; nf < 4; ++nf) {
      const int gn = bn + wn * 64 + nf * 16 + l16;
      const float bv = bias ? bias[gn] : 0.0f;
      for (int r = 0; r < 8; ++r) {
        const int gm = bm + wm * 32 + mf * 16 + half * 8 + r;
        C[(size_t)gm * Nn + gn] = acc[mf][nf][r] + bv;
      }
    }
}

// --------------------------------------------------- dual-B gated FFN GEMM
// Hb[M,N] bf16 = gelu(A*W1^T + b1) * (A*W3^T + b3). Async-LDS double buffer.
__global__ __launch_bounds__(256)
void k_ffn_gate(const unsigned short* __restrict__ A,
                const unsigned short* __restrict__ B1w,
                const unsigned short* __restrict__ B3w,
                const float* __restrict__ bias1,
                const float* __restrict__ bias3,
                unsigned short* __restrict__ Hb,
                int Mm, int Nn, int Kk) {
  __shared__ __attribute__((aligned(16))) unsigned short As [2][TM][TK];
  __shared__ __attribute__((aligned(16))) unsigned short B1s[2][64][TK];
  __shared__ __attribute__((aligned(16))) unsigned short B3s[2][64][TK];
  const int tid  = threadIdx.x;
  const int lane = tid & 31, wid = tid >> 5;
  const int half = lane >> 4, l16 = lane & 15;
  const int wm = wid & 3, wn = wid >> 2;
  const int bm = blockIdx.y * TM, bn = blockIdx.x * 64;
  const int lr = tid >> 1, lc = (tid & 1) << 4;
  const int br = tid >> 2, bc = (tid & 3) << 3;

  auto stage = [&](int buf, int k0) {
    const unsigned short* ga = &A[(size_t)(bm + lr) * Kk + k0 + lc];
    async_g2l_b128(&As[buf][lr][lc],     ga);
    async_g2l_b128(&As[buf][lr][lc + 8], ga + 8);
    async_g2l_b128(&B1s[buf][br][bc], &B1w[(size_t)(bn + br) * Kk + k0 + bc]);
    async_g2l_b128(&B3s[buf][br][bc], &B3w[(size_t)(bn + br) * Kk + k0 + bc]);
  };

  const v8f vzero = {0.f,0.f,0.f,0.f,0.f,0.f,0.f,0.f};
  v8f a1[2][2], a3[2][2];
  for (int i = 0; i < 2; ++i) for (int j = 0; j < 2; ++j) { a1[i][j] = vzero; a3[i][j] = vzero; }

  stage(0, 0);
  wait_asynccnt0();
  __syncthreads();

  for (int k0 = 0; k0 < Kk; k0 += TK) {
    const int cur = (k0 >> 5) & 1;
    if (k0 + TK < Kk) stage(cur ^ 1, k0 + TK);

    v16bf af[2], b1f[2], b3f[2];
    for (int mf = 0; mf < 2; ++mf) {
      const int m = wm * 32 + mf * 16 + l16;
      v8u lo = *(const v8u*)&As[cur][m][half * 8];
      v8u hi = *(const v8u*)&As[cur][m][16 + half * 8];
      v16u t;
      for (int i = 0; i < 8; ++i) { t[i] = lo[i]; t[i + 8] = hi[i]; }
      af[mf] = __builtin_bit_cast(v16bf, t);
    }
    for (int nf = 0; nf < 2; ++nf) {
      const int n = wn * 32 + nf * 16 + l16;
      { v8u lo = *(const v8u*)&B1s[cur][n][half * 8];
        v8u hi = *(const v8u*)&B1s[cur][n][16 + half * 8];
        v16u t; for (int i = 0; i < 8; ++i) { t[i] = lo[i]; t[i + 8] = hi[i]; }
        b1f[nf] = __builtin_bit_cast(v16bf, t); }
      { v8u lo = *(const v8u*)&B3s[cur][n][half * 8];
        v8u hi = *(const v8u*)&B3s[cur][n][16 + half * 8];
        v16u t; for (int i = 0; i < 8; ++i) { t[i] = lo[i]; t[i + 8] = hi[i]; }
        b3f[nf] = __builtin_bit_cast(v16bf, t); }
    }
    for (int mf = 0; mf < 2; ++mf)
      for (int nf = 0; nf < 2; ++nf) {
        a1[mf][nf] = __builtin_amdgcn_wmma_f32_16x16x32_bf16(
            false, af[mf], false, b1f[nf], (short)0, a1[mf][nf], false, false);
        a3[mf][nf] = __builtin_amdgcn_wmma_f32_16x16x32_bf16(
            false, af[mf], false, b3f[nf], (short)0, a3[mf][nf], false, false);
      }

    wait_asynccnt0();
    __syncthreads();
  }

  for (int mf = 0; mf < 2; ++mf)
    for (int nf = 0; nf < 2; ++nf) {
      const int gn = bn + wn * 32 + nf * 16 + l16;
      const float bb1 = bias1[gn], bb3 = bias3[gn];
      for (int r = 0; r < 8; ++r) {
        const int gm = bm + wm * 32 + mf * 16 + half * 8 + r;
        float p1 = a1[mf][nf][r] + bb1;
        float p3 = a3[mf][nf][r] + bb3;
        float ge = 0.5f * p1 * (1.0f + erff(p1 * 0.7071067811865476f));  // exact gelu
        Hb[(size_t)gm * Nn + gn] = f2bf(ge * p3);
      }
    }
}

// ------------------------------------------------------------ infini-attention
// One workgroup per (b,h). 128x128 memory + delta accumulator live in LDS across
// the 8-segment scan. fp32 WMMA (16x16x4) for QK^T, PV, sq@mem, sk@mem.
// K/V tiles staged with async global->LDS copies.
__global__ __launch_bounds__(256)
void k_attn(const float* __restrict__ Qf, const float* __restrict__ Kf,
            const float* __restrict__ Vf, float* __restrict__ Outf,
            const float* __restrict__ beta) {
  extern __shared__ float sm_[];
  float* mem  = sm_;                 // 128*128
  float* dmem = mem  + 128 * 128;    // 128*128
  float* zvec = dmem + 128 * 128;    // 128
  float* dz   = zvec + 128;          // 128
  float* qt   = dz   + 128;          // 64*128
  float* sqt  = qt   + 64 * 128;     // 64*128 (sq, reused for sk in phase 2)
  float* kt   = sqt  + 64 * 128;     // 64*128
  float* vt   = kt   + 64 * 128;     // 64*128
  float* St   = vt   + 64 * 128;     // 64*64
  float* accS = St   + 64 * 64;      // 64*128 (flash acc / v_delta)
  float* mrow = accS + 64 * 128;     // 64
  float* lrow = mrow + 64;           // 64
  float* frow = lrow + 64;           // 64
  float* drow = frow + 64;           // 64

  const int tid  = threadIdx.x;
  const int lane = tid & 31, wid = tid >> 5;
  const int half = lane >> 4, l16 = lane & 15;
  const int b = blockIdx.x >> 4;
  const int h = blockIdx.x & 15;
  const float g = 1.0f / (1.0f + __expf(-beta[0]));
  const float scale = 0.08838834764831845f;  // 1/sqrt(DH)
  const size_t baseRow = (size_t)b * S_;
  const size_t colOff  = (size_t)h * DH_;
  const v8f vzero = {0.f,0.f,0.f,0.f,0.f,0.f,0.f,0.f};

  for (int i = tid; i < 128 * 128; i += 256) mem[i] = 0.0f;
  if (tid < 128) zvec[tid] = 0.0f;
  __syncthreads();

  for (int seg = 0; seg < NSEG_; ++seg) {
    // -------- Phase 1: outputs (use carried mem/z) --------
    for (int qi = 0; qi < 8; ++qi) {
      __syncthreads();
      for (int idx = tid; idx < 64 * 128; idx += 256) {
        int rr = idx >> 7, dd = idx & 127;
        size_t grow = baseRow + (size_t)seg * SEG_ + qi * 64 + rr;
        float qv = Qf[grow * E_ + colOff + dd];
        qt[idx]   = qv;
        sqt[idx]  = qv > 0.f ? qv + 1.f : __expf(qv);   // elu(q)+1
        accS[idx] = 0.0f;
      }
      if (tid < 64) { mrow[tid] = -1e30f; lrow[tid] = 0.0f; }
      __syncthreads();
      if (tid < 64) {
        float s = 0.f;
        for (int d = 0; d < 128; ++d) s += sqt[tid * 128 + d] * zvec[d];
        drow[tid] = s + 1e-6f;
      }
      // flash loop over causal key tiles
      for (int kj = 0; kj <= qi; ++kj) {
        __syncthreads();
        // async copy K/V tiles (pure copies -> ASYNCcnt path)
        for (int slot = tid; slot < 64 * 32; slot += 256) {   // 4-float chunks
          int rr = slot >> 5, cc = (slot & 31) << 2;
          size_t grow = baseRow + (size_t)seg * SEG_ + kj * 64 + rr;
          async_g2l_b128(&kt[rr * 128 + cc], &Kf[grow * E_ + colOff + cc]);
          async_g2l_b128(&vt[rr * 128 + cc], &Vf[grow * E_ + colOff + cc]);
        }
        wait_asynccnt0();
        __syncthreads();
        {   // scores 64x64 = q @ k^T  (f32 WMMA, K=4, 32 steps)
          const int fm = wid & 3;
          const int fn0 = (wid >> 2) * 2;
          for (int fi = 0; fi < 2; ++fi) {
            const int fn = fn0 + fi;
            v8f sa = vzero;
            const int mi = fm * 16 + l16;
            const int ni = fn * 16 + l16;
            for (int kk = 0; kk < 128; kk += 4) {
              v2f a2 = *(const v2f*)&qt[mi * 128 + kk + 2 * half];
              v2f b2 = *(const v2f*)&kt[ni * 128 + kk + 2 * half];
              sa = __builtin_amdgcn_wmma_f32_16x16x4_f32(false, a2, false, b2,
                                                         (short)0, sa, false, false);
            }
            for (int r = 0; r < 8; ++r) {
              int smr = fm * 16 + half * 8 + r;
              int snc = fn * 16 + l16;
              float sv = sa[r] * scale;
              if (kj == qi && snc > smr) sv = -1e30f;    // causal
              St[smr * 64 + snc] = sv;
            }
          }
        }
        __syncthreads();
        if (tid < 64) {  // online softmax row update
          float mx = mrow[tid];
          for (int c = 0; c < 64; ++c) mx = fmaxf(mx, St[tid * 64 + c]);
          float f = __expf(mrow[tid] - mx);
          float sum = 0.f;
          for (int c = 0; c < 64; ++c) {
            float p = __expf(St[tid * 64 + c] - mx);
            St[tid * 64 + c] = p;
            sum += p;
          }
          lrow[tid] = lrow[tid] * f + sum;
          mrow[tid] = mx;
          frow[tid] = f;
        }
        __syncthreads();
        for (int idx = tid; idx < 64 * 128; idx += 256) accS[idx] *= frow[idx >> 7];
        __syncthreads();
        {   // accS += P @ V  (64x64 @ 64x128)
          const int fm = wid & 3;
          const int fn0 = (wid >> 2) * 4;
          for (int fi = 0; fi < 4; ++fi) {
            const int fn = fn0 + fi;
            v8f pv = vzero;
            const int mi = fm * 16 + l16;
            const int nc = fn * 16 + l16;
            for (int kk = 0; kk < 64; kk += 4) {
              v2f a2 = *(const v2f*)&St[mi * 64 + kk + 2 * half];
              v2f b2;
              b2.x = vt[(kk + 2 * half    ) * 128 + nc];
              b2.y = vt[(kk + 2 * half + 1) * 128 + nc];
              pv = __builtin_amdgcn_wmma_f32_16x16x4_f32(false, a2, false, b2,
                                                         (short)0, pv, false, false);
            }
            for (int r = 0; r < 8; ++r)
              accS[(fm * 16 + half * 8 + r) * 128 + nc] += pv[r];
          }
        }
      }  // kj
      __syncthreads();
      {   // A_mem = sq @ mem, combine, write out
        const int fm = wid & 3;
        const int fn0 = (wid >> 2) * 4;
        for (int fi = 0; fi < 4; ++fi) {
          const int fn = fn0 + fi;
          v8f am = vzero;
          const int mi = fm * 16 + l16;
          const int nc = fn * 16 + l16;
          for (int kk = 0; kk < 128; kk += 4) {
            v2f a2 = *(const v2f*)&sqt[mi * 128 + kk + 2 * half];
            v2f b2;
            b2.x = mem[(kk + 2 * half    ) * 128 + nc];
            b2.y = mem[(kk + 2 * half + 1) * 128 + nc];
            am = __builtin_amdgcn_wmma_f32_16x16x4_f32(false, a2, false, b2,
                                                       (short)0, am, false, false);
          }
          for (int r = 0; r < 8; ++r) {
            int smr = fm * 16 + half * 8 + r;
            size_t grow = baseRow + (size_t)seg * SEG_ + qi * 64 + smr;
            float adot = accS[smr * 128 + nc] / lrow[smr];
            float amem = am[r] / drow[smr];
            Outf[grow * E_ + colOff + nc] = g * amem + (1.f - g) * adot;
          }
        }
      }
    }  // qi

    // -------- Phase 2: delta-rule memory update (all tiles vs carried mem/z) --------
    __syncthreads();
    for (int i = tid; i < 128 * 128; i += 256) dmem[i] = 0.f;
    if (tid < 128) dz[tid] = 0.f;
    for (int ki = 0; ki < 8; ++ki) {
      __syncthreads();
      // async V tile; K loaded through VALU for elu transform
      for (int slot = tid; slot < 64 * 32; slot += 256) {
        int rr = slot >> 5, cc = (slot & 31) << 2;
        size_t grow = baseRow + (size_t)seg * SEG_ + ki * 64 + rr;
        async_g2l_b128(&vt[rr * 128 + cc], &Vf[grow * E_ + colOff + cc]);
      }
      for (int idx = tid; idx < 64 * 128; idx += 256) {
        int rr = idx >> 7, dd = idx & 127;
        size_t grow = baseRow + (size_t)seg * SEG_ + ki * 64 + rr;
        float kv = Kf[grow * E_ + colOff + dd];
        sqt[idx] = kv > 0.f ? kv + 1.f : __expf(kv);   // sk
      }
      wait_asynccnt0();
      __syncthreads();
      if (tid < 64) {
        float s = 0.f;
        for (int d = 0; d < 128; ++d) s += sqt[tid * 128 + d] * zvec[d];
        drow[tid] = s + 1e-6f;
      }
      __syncthreads();
      {   // v_delta = v - (sk @ mem) / (sk @ z)
        const int fm = wid & 3;
        const int fn0 = (wid >> 2) * 4;
        for (int fi = 0; fi < 4; ++fi) {
          const int fn = fn0 + fi;
          v8f rm = vzero;
          const int mi = fm * 16 + l16;
          const int nc = fn * 16 + l16;
          for (int kk = 0; kk < 128; kk += 4) {
            v2f a2 = *(const v2f*)&sqt[mi * 128 + kk + 2 * half];
            v2f b2;
            b2.x = mem[(kk + 2 * half    ) * 128 + nc];
            b2.y = mem[(kk + 2 * half + 1) * 128 + nc];
            rm = __builtin_amdgcn_wmma_f32_16x16x4_f32(false, a2, false, b2,
                                                       (short)0, rm, false, false);
          }
          for (int r = 0; r < 8; ++r) {
            int smr = fm * 16 + half * 8 + r;
            accS[smr * 128 + nc] = vt[smr * 128 + nc] - rm[r] / drow[smr];
          }
        }
      }
      __syncthreads();
      // dmem += sk^T @ v_delta ; dz += colsum(sk)
      for (int idx = tid; idx < 128 * 128; idx += 256) {
        int d = idx >> 7, e = idx & 127;
        float s = 0.f;
        for (int t2 = 0; t2 < 64; ++t2) s += sqt[t2 * 128 + d] * accS[t2 * 128 + e];
        dmem[idx] += s;
      }
      if (tid < 128) {
        float s = 0.f;
        for (int t2 = 0; t2 < 64; ++t2) s += sqt[t2 * 128 + tid];
        dz[tid] += s;
      }
    }  // ki
    __syncthreads();
    for (int i = tid; i < 128 * 128; i += 256) mem[i] += dmem[i];
    if (tid < 128) zvec[tid] += dz[tid];
    __syncthreads();
  }  // seg
}

// ------------------------------------------------- residual + RMSNorm (+bf16)
__global__ __launch_bounds__(256)
void k_addnorm(const float* __restrict__ resid, const float* __restrict__ pre,
               const float* __restrict__ w, float* __restrict__ yout,
               unsigned short* __restrict__ ybf) {
  __shared__ float red[256];
  const int row = blockIdx.x;
  const int t = threadIdx.x;
  float v[8];
  float ss = 0.f;
  for (int j = 0; j < 8; ++j) {
    v[j] = pre[(size_t)row * E_ + t + j * 256];
    ss += v[j] * v[j];
  }
  red[t] = ss;
  __syncthreads();
  for (int s2 = 128; s2 > 0; s2 >>= 1) {
    if (t < s2) red[t] += red[t + s2];
    __syncthreads();
  }
  const float rms = rsqrtf(red[0] * (1.0f / E_) + 1e-5f);
  for (int j = 0; j < 8; ++j) {
    int c = t + j * 256;
    float o = resid[(size_t)row * E_ + c] + v[j] * rms * w[c];
    yout[(size_t)row * E_ + c] = o;
    if (ybf) ybf[(size_t)row * E_ + c] = f2bf(o);
  }
}

// ---------------------------------------------------------------- launcher
extern "C" void kernel_launch(void* const* d_in, const int* in_sizes, int n_in,
                              void* d_out, int out_size, void* d_ws, size_t ws_size,
                              hipStream_t stream) {
  (void)in_sizes; (void)n_in; (void)out_size; (void)ws_size;
  const float* x    = (const float*)d_in[0];
  const float* Wq   = (const float*)d_in[1];
  const float* bq   = (const float*)d_in[2];
  const float* Wk   = (const float*)d_in[3];
  const float* bk   = (const float*)d_in[4];
  const float* Wv   = (const float*)d_in[5];
  const float* bv   = (const float*)d_in[6];
  const float* Wo   = (const float*)d_in[7];
  const float* bo   = (const float*)d_in[8];
  const float* beta = (const float*)d_in[9];
  const float* anw  = (const float*)d_in[10];
  const float* fnw  = (const float*)d_in[11];
  const float* W1   = (const float*)d_in[12];
  const float* b1   = (const float*)d_in[13];
  const float* W2   = (const float*)d_in[14];
  const float* b2   = (const float*)d_in[15];
  const float* W3   = (const float*)d_in[16];
  const float* b3   = (const float*)d_in[17];
  float* out = (float*)d_out;

  const size_t Mrow = (size_t)B_ * S_;   // 16384
  char* ws = (char*)d_ws;
  size_t off = 0;
  auto alloc = [&](size_t bytes) -> void* {
    void* p = ws + off;
    off = (off + bytes + 255) & ~(size_t)255;
    return p;
  };

  unsigned short* xb    = (unsigned short*)alloc(Mrow * E_ * 2);
  unsigned short* wqb   = (unsigned short*)alloc((size_t)E_ * E_ * 2);
  unsigned short* wkb   = (unsigned short*)alloc((size_t)E_ * E_ * 2);
  unsigned short* wvb   = (unsigned short*)alloc((size_t)E_ * E_ * 2);
  unsigned short* wob   = (unsigned short*)alloc((size_t)E_ * E_ * 2);
  unsigned short* w1b   = (unsigned short*)alloc((size_t)FE_ * E_ * 2);
  unsigned short* w3b   = (unsigned short*)alloc((size_t)FE_ * E_ * 2);
  unsigned short* w2b   = (unsigned short*)alloc((size_t)E_ * FE_ * 2);
  float*          Qfb   = (float*)alloc(Mrow * E_ * 4);
  float*          Kfb   = (float*)alloc(Mrow * E_ * 4);
  float*          Vfb   = (float*)alloc(Mrow * E_ * 4);
  float*          attnf = (float*)alloc(Mrow * E_ * 4);
  unsigned short* attnb = (unsigned short*)alloc(Mrow * E_ * 2);
  float*          projf = (float*)alloc(Mrow * E_ * 4);
  float*          yf    = (float*)alloc(Mrow * E_ * 4);
  unsigned short* yb    = (unsigned short*)alloc(Mrow * E_ * 2);
  unsigned short* hb    = (unsigned short*)alloc(Mrow * FE_ * 2);
  float*          f2v   = (float*)alloc(Mrow * E_ * 4);

  // 1. bf16 casts (weights stay L2-resident at 134 MB total)
  k_cast_bf16<<<2048, 256, 0, stream>>>(x,  xb,  (long long)(Mrow * E_));
  k_cast_bf16<<<1024, 256, 0, stream>>>(Wq, wqb, (long long)E_ * E_);
  k_cast_bf16<<<1024, 256, 0, stream>>>(Wk, wkb, (long long)E_ * E_);
  k_cast_bf16<<<1024, 256, 0, stream>>>(Wv, wvb, (long long)E_ * E_);
  k_cast_bf16<<<1024, 256, 0, stream>>>(Wo, wob, (long long)E_ * E_);
  k_cast_bf16<<<2048, 256, 0, stream>>>(W1, w1b, (long long)FE_ * E_);
  k_cast_bf16<<<2048, 256, 0, stream>>>(W2, w2b, (long long)E_ * FE_);
  k_cast_bf16<<<2048, 256, 0, stream>>>(W3, w3b, (long long)FE_ * E_);

  // 2. QKV projections
  dim3 blk(256);
  dim3 gqkv(E_ / TN, Mrow / TM);  // (16, 128)
  k_gemm_bias<<<gqkv, blk, 0, stream>>>(xb, wqb, bq, Qfb, (int)Mrow, E_, E_);
  k_gemm_bias<<<gqkv, blk, 0, stream>>>(xb, wkb, bk, Kfb, (int)Mrow, E_, E_);
  k_gemm_bias<<<gqkv, blk, 0, stream>>>(xb, wvb, bv, Vfb, (int)Mrow, E_, E_);

  // 3. infini-attention (B*H workgroups, 8-segment scan)
  const size_t attnShm =
      (2 * 128 * 128 + 2 * 128 + 4 * 64 * 128 + 64 * 64 + 64 * 128 + 4 * 64) *
      sizeof(float);  // ~313 KB
  k_attn<<<B_ * H_, blk, attnShm, stream>>>(Qfb, Kfb, Vfb, attnf, beta);

  // 4. output projection + residual/RMSNorm
  k_cast_bf16<<<2048, 256, 0, stream>>>(attnf, attnb, (long long)(Mrow * E_));
  k_gemm_bias<<<gqkv, blk, 0, stream>>>(attnb, wob, bo, projf, (int)Mrow, E_, E_);
  k_addnorm<<<(unsigned)Mrow, blk, 0, stream>>>(x, projf, anw, yf, yb);

  // 5. gated FFN
  dim3 gffn(FE_ / 64, Mrow / TM);  // (128, 128)
  k_ffn_gate<<<gffn, blk, 0, stream>>>(yb, w1b, w3b, b1, b3, hb, (int)Mrow, FE_, E_);
  dim3 gw2(E_ / TN, Mrow / TM);
  k_gemm_bias<<<gw2, blk, 0, stream>>>(hb, w2b, b2, f2v, (int)Mrow, E_, FE_);
  k_addnorm<<<(unsigned)Mrow, blk, 0, stream>>>(yf, f2v, fnw, out, (unsigned short*)nullptr);
}
